// LSTM_5385888989345
// MI455X (gfx1250) — matrix operand order repaired and checked by reading, not compiled, not run on hardware
//
#include <hip/hip_runtime.h>
#include <hip/hip_bf16.h>

// Problem constants (reference: V,D,H,P = 30000,2048,512,196; B=32; T=19)
#define BATCH  32
#define PPIX   196
#define DDIM   2048
#define HDIM   512
#define VOCAB  30000
#define VPAD   30080   // VOCAB padded to multiple of 128 (235 slabs)
#define TSTEPS 19
#define CAPLEN 20
#define XDIM   3072    // [emb(H) | gate*context(D) | h(H)]

typedef __bf16 bf16_t;
typedef __attribute__((ext_vector_type(16))) __bf16 bf16x16;
typedef __attribute__((ext_vector_type(8)))  float  f32x8;

__device__ __forceinline__ float sigmoidf_(float x) { return 1.0f / (1.0f + __expf(-x)); }

// CDNA5 async memory->LDS copies (tracked by ASYNCcnt). lds_off = low 32 bits
// of the flat shared address == LDS byte offset (ISA 10.2).
__device__ __forceinline__ void async_copy_b128(unsigned lds_off, const void* gptr) {
    asm volatile("global_load_async_to_lds_b128 %0, %1, off"
                 :: "v"(lds_off), "v"((unsigned long long)(uintptr_t)gptr)
                 : "memory");
}
__device__ __forceinline__ void async_copy_b32(unsigned lds_off, const void* gptr) {
    asm volatile("global_load_async_to_lds_b32 %0, %1, off"
                 :: "v"(lds_off), "v"((unsigned long long)(uintptr_t)gptr)
                 : "memory");
}

// ---------------------------------------------------------------------------
// bf16 WMMA GEMM: C[M,N](f32) = act(A[M,K] @ B[K,N] + bias)
//  - B pre-packed in global memory: Bp[k/2][npad] u32 = {B[k][n], B[k+1][n]},
//    npad multiple of 128 (zero padded) -> branch-free K loop; the LDS image
//    is directly the WMMA B fragment (lane n, VGPR v -> Bs[hi*8+v][n]).
//  - Staging: GLOBAL_LOAD_ASYNC_TO_LDS_{B128,B32}, uniform 3 async ops per
//    thread per tile, 3-deep LDS pipeline, one barrier + one constant
//    s_wait_asynccnt per K tile (last tile peeled).
//  - One wave computes one 16x16 D tile via v_wmma_f32_16x16x32_bf16.
//  act: 0 = none, 1 = tanh, 2 = sigmoid
// ---------------------------------------------------------------------------
__global__ __launch_bounds__(256)
void wmma_gemm_bf16(const bf16_t* __restrict__ A, int lda,
                    const unsigned* __restrict__ Bp, int npad,
                    const float* __restrict__ bias,
                    float* __restrict__ C, long ldc,
                    int M, int K, int N, int act)
{
    // A: 16x32 bf16 rows padded to 40 elems (80B: 16B-aligned, conflict-free)
    // B: 16 k-pair rows x 128 u32 padded to 132 (528B: 16B-aligned; +8 rows
    //    shift banks by 32 -> the two lane-halves never collide)
    __shared__ bf16_t   As[3][16][40];
    __shared__ unsigned Bs[3][16][132];

    const int tid  = threadIdx.x;
    const int lane = tid & 31;
    const int wave = tid >> 5;
    const int tm   = blockIdx.y;          // 16-row M tile
    const int nb   = blockIdx.x * 128;    // 128-col N slab
    const int wn   = wave * 16;           // wave's 16-col tile in slab

    union Acc { f32x8 v; float f[8]; } acc;
#pragma unroll
    for (int i = 0; i < 8; ++i) acc.f[i] = 0.0f;

    // staging assignments: every thread issues exactly 3 async ops per tile
    const int brow = wave;                // B k2-rows: brow, brow+8 (b128 each)
    const int bcu  = lane * 4;            // u32 column (16B chunk)
    const int arow = tid >> 4;            // A row 0..15
    const int ac2  = (tid & 15) * 2;      // A col pair (b32)

    const unsigned offB1 = (unsigned)(brow * 132 + bcu) * 4u;
    const unsigned offB2 = (unsigned)((brow + 8) * 132 + bcu) * 4u;
    const unsigned offA  = (unsigned)(arow * 40 + ac2) * 2u;

    // incrementally advanced global pointers (no per-iter 64-bit muls)
    const unsigned* gB1 = Bp + (long)brow * npad + nb + bcu;
    const unsigned* gB2 = gB1 + (long)8 * npad;
    const bf16_t*   gA  = A + (long)(tm * 16 + arow) * lda + ac2;
    const long stepB = (long)16 * npad;

    auto issue = [&](int buf) {
        const unsigned dB = (unsigned)(uintptr_t)&Bs[buf][0][0];
        const unsigned dA = (unsigned)(uintptr_t)&As[buf][0][0];
        async_copy_b128(dB + offB1, gB1);
        async_copy_b128(dB + offB2, gB2);
        async_copy_b32 (dA + offA,  gA);
        gB1 += stepB; gB2 += stepB; gA += 32;
    };

    const int mrow = lane & 15;
    const int hi   = lane >> 4;
    auto compute = [&](int buf) {
        union Frag { bf16x16 v; unsigned u[8]; } fa, fb;
#pragma unroll
        for (int v = 0; v < 8; ++v) {
            // A 16x32 bf16 fragment: VGPR v K-pair base = hi*16+v*2
            const int ka = (hi << 4) + (v << 1);
            fa.u[v] = *reinterpret_cast<const unsigned*>(&As[buf][mrow][ka]);
            // B 32x16 fragment: lanes 0-15 K=0..15, lanes 16-31 K=16..31;
            // packed u32 = K pair (2k2, 2k2+1)
            fb.u[v] = Bs[buf][hi * 8 + v][wn + mrow];
        }
        acc.v = __builtin_amdgcn_wmma_f32_16x16x32_bf16(
                    /*neg_a=*/false, fa.v, /*neg_b=*/false, fb.v,
                    /*c_mod=*/(short)0, acc.v, /*reuse_a=*/false, /*reuse_b=*/false);
    };

    const int ntiles = K >> 5;
    issue(0);                      // pre-issue tile 0 (batch size 3, uniform)
    int cbuf = 0, ibuf = 1;
    for (int i = 0; i < ntiles - 1; ++i) {
        issue(ibuf);
        // retire current tile's batch: allowed outstanding = next batch (3)
        asm volatile("s_wait_asynccnt 0x3" ::: "memory");
        __syncthreads();           // 3 buffers -> one barrier per tile suffices
        compute(cbuf);
        cbuf = (cbuf == 2) ? 0 : cbuf + 1;
        ibuf = (ibuf == 2) ? 0 : ibuf + 1;
    }
    asm volatile("s_wait_asynccnt 0x0" ::: "memory");   // last tile (peeled)
    __syncthreads();
    compute(cbuf);

    // epilogue: bias + activation + store (C/D layout: M = r + 8*hi)
    const int col = nb + wn + (lane & 15);
    if (col < N) {
#pragma unroll
        for (int r = 0; r < 8; ++r) {
            const int row = tm * 16 + hi * 8 + r;
            if (row < M) {
                float v = acc.f[r] + (bias ? bias[col] : 0.0f);
                if (act == 1)      v = tanhf(v);
                else if (act == 2) v = sigmoidf_(v);
                C[(long)row * ldc + col] = v;
            }
        }
    }
}

// ---------------------------------------------------------------------------
// Fused soft-attention per batch element: e -> softmax -> alpha -> context
// ---------------------------------------------------------------------------
__global__ __launch_bounds__(256)
void attention_kernel(const float* __restrict__ att_img,   // [B,P,H]
                      const float* __restrict__ hW,        // [B,H]
                      const float* __restrict__ v_att,     // [H]
                      const float* __restrict__ b_v_att,   // [1]
                      const bf16_t* __restrict__ img_bf,   // [B,P,D]
                      float* __restrict__ context,         // [B,D]
                      float* __restrict__ alphas,          // [B,T,P] slab of d_out
                      int t)
{
    __shared__ float sh_h[HDIM];
    __shared__ float sh_v[HDIM];
    __shared__ float sh_e[PPIX];
    __shared__ float sh_mx, sh_sm;

    const int b    = blockIdx.x;
    const int tid  = threadIdx.x;
    const int lane = tid & 31;
    const int wave = tid >> 5;

    for (int i = tid; i < HDIM; i += 256) {
        sh_h[i] = hW[b * HDIM + i];
        sh_v[i] = v_att[i];
    }
    __syncthreads();

    const float bv = b_v_att[0];
    for (int p = wave; p < PPIX; p += 8) {
        const float* row = &att_img[((long)b * PPIX + p) * HDIM];
        float s = 0.0f;
        for (int h = lane; h < HDIM; h += 32)
            s += tanhf(row[h] + sh_h[h]) * sh_v[h];
#pragma unroll
        for (int off = 16; off > 0; off >>= 1) s += __shfl_xor(s, off, 32);
        if (lane == 0) sh_e[p] = s + bv;
    }
    __syncthreads();

    if (wave == 0) {
        float mx = -3.4e38f;
        for (int p = lane; p < PPIX; p += 32) mx = fmaxf(mx, sh_e[p]);
#pragma unroll
        for (int off = 16; off > 0; off >>= 1) mx = fmaxf(mx, __shfl_xor(mx, off, 32));
        float sm = 0.0f;
        for (int p = lane; p < PPIX; p += 32) sm += __expf(sh_e[p] - mx);
#pragma unroll
        for (int off = 16; off > 0; off >>= 1) sm += __shfl_xor(sm, off, 32);
        if (lane == 0) { sh_mx = mx; sh_sm = sm; }
    }
    __syncthreads();

    const float mx = sh_mx, inv = 1.0f / sh_sm;
    for (int p = tid; p < PPIX; p += 256) {
        const float a = __expf(sh_e[p] - mx) * inv;
        sh_e[p] = a;
        alphas[((long)b * TSTEPS + t) * PPIX + p] = a;
    }
    __syncthreads();

    // context[d] = sum_p alpha[p]*img[b,p,d]; 8 d per thread, b128 loads
    const int d0 = tid * 8;
    float acc[8];
#pragma unroll
    for (int j = 0; j < 8; ++j) acc[j] = 0.0f;
    for (int p = 0; p < PPIX; ++p) {
        const float a = sh_e[p];
        bf16_t tmp[8];
        *reinterpret_cast<uint4*>(tmp) =
            *reinterpret_cast<const uint4*>(&img_bf[((long)b * PPIX + p) * DDIM + d0]);
#pragma unroll
        for (int j = 0; j < 8; ++j) acc[j] += a * (float)tmp[j];
    }
#pragma unroll
    for (int j = 0; j < 8; ++j) context[b * DDIM + d0 + j] = acc[j];
}

// ---------------------------------------------------------------------------
// Helper kernels
// ---------------------------------------------------------------------------
__global__ void cvt_f32_bf16(const float* __restrict__ s, bf16_t* __restrict__ d, long n) {
    for (long i = blockIdx.x * 256L + threadIdx.x; i < n; i += (long)gridDim.x * 256L)
        d[i] = (bf16_t)s[i];
}

// Pack weight [Ktot, Nsrc] f32 (optionally split row-wise across s1/s2 at rows1)
// into K-pair-interleaved bf16 u32 image [Ktot/2, Npad], zero-padding cols.
__global__ void cvt_pack_bf16(const float* __restrict__ s1, int rows1,
                              const float* __restrict__ s2,
                              unsigned* __restrict__ d,
                              int Ktot, int Nsrc, int Npad, long n) {
    for (long i = blockIdx.x * 256L + threadIdx.x; i < n; i += (long)gridDim.x * 256L) {
        const int k2  = (int)(i / Npad);
        const int col = (int)(i % Npad);
        unsigned out = 0;
        if (col < Nsrc) {
            const int r0 = 2 * k2, r1 = 2 * k2 + 1;
            const float f0 = (r0 < rows1) ? s1[(long)r0 * Nsrc + col]
                                          : s2[(long)(r0 - rows1) * Nsrc + col];
            const float f1 = (r1 < rows1) ? s1[(long)r1 * Nsrc + col]
                                          : s2[(long)(r1 - rows1) * Nsrc + col];
            union { bf16_t h[2]; unsigned u; } p;
            p.h[0] = (bf16_t)f0; p.h[1] = (bf16_t)f1;
            out = p.u;
        }
        d[i] = out;
    }
}

__global__ void avgpool_kernel(const float* __restrict__ img, bf16_t* __restrict__ avg_b) {
    const int idx = blockIdx.x * 256 + threadIdx.x;   // B*D threads
    const int b = idx >> 11, d = idx & (DDIM - 1);
    float s = 0.0f;
    for (int p = 0; p < PPIX; ++p) s += img[((long)b * PPIX + p) * DDIM + d];
    avg_b[idx] = (bf16_t)(s * (1.0f / PPIX));
}

__global__ void gather_emb(const int* __restrict__ captions, const float* __restrict__ E,
                           bf16_t* __restrict__ emb, long n) {
    for (long i = blockIdx.x * 256L + threadIdx.x; i < n; i += (long)gridDim.x * 256L) {
        const int bi = (int)(i / (CAPLEN * HDIM));
        const int r  = (int)(i % (CAPLEN * HDIM));
        const int ti = r / HDIM, ci = r % HDIM;
        const int tok = captions[bi * CAPLEN + ti];
        emb[i] = (bf16_t)E[(long)tok * HDIM + ci];
    }
}

__global__ void add_bias2(const float* __restrict__ a, const float* __restrict__ b,
                          float* __restrict__ o, int n) {
    const int i = blockIdx.x * 256 + threadIdx.x;
    if (i < n) o[i] = a[i] + b[i];
}

__global__ void build_x(const bf16_t* __restrict__ emb, const float* __restrict__ gate,
                        const float* __restrict__ context, const bf16_t* __restrict__ h_bf,
                        bf16_t* __restrict__ x, int t) {
    const int idx = blockIdx.x * 256 + threadIdx.x;   // B*XDIM
    const int b = idx / XDIM, j = idx % XDIM;
    bf16_t v;
    if (j < HDIM)             v = emb[((long)b * CAPLEN + t) * HDIM + j];
    else if (j < HDIM + DDIM) {
        const int d = j - HDIM;
        v = (bf16_t)(gate[b * DDIM + d] * context[b * DDIM + d]);
    } else                     v = h_bf[b * HDIM + (j - HDIM - DDIM)];
    x[idx] = v;
}

__global__ void lstm_pointwise(const float* __restrict__ gates, float* __restrict__ c,
                               float* __restrict__ h, bf16_t* __restrict__ h_bf) {
    const int idx = blockIdx.x * 256 + threadIdx.x;   // B*H
    const int b = idx >> 9, j = idx & (HDIM - 1);
    const float* g = &gates[(long)b * 4 * HDIM];
    const float iv = sigmoidf_(g[j]);
    const float fv = sigmoidf_(g[HDIM + j]);
    const float gv = tanhf(g[2 * HDIM + j]);
    const float ov = sigmoidf_(g[3 * HDIM + j]);
    const float cn = fv * c[idx] + iv * gv;
    const float hn = ov * tanhf(cn);
    c[idx] = cn; h[idx] = hn; h_bf[idx] = (bf16_t)hn;
}

// ---------------------------------------------------------------------------
// Host-side orchestration
// ---------------------------------------------------------------------------
extern "C" void kernel_launch(void* const* d_in, const int* in_sizes, int n_in,
                              void* d_out, int out_size, void* d_ws, size_t ws_size,
                              hipStream_t stream) {
    const float* img       = (const float*)d_in[0];
    const int*   captions  = (const int*)  d_in[1];
    const float* Wih       = (const float*)d_in[2];
    const float* Whh       = (const float*)d_in[3];
    const float* bih       = (const float*)d_in[4];
    const float* bhh       = (const float*)d_in[5];
    const float* W_init_h  = (const float*)d_in[6];
    const float* b_init_h  = (const float*)d_in[7];
    const float* W_init_c  = (const float*)d_in[8];
    const float* b_init_c  = (const float*)d_in[9];
    const float* W_fbeta   = (const float*)d_in[10];
    const float* b_fbeta   = (const float*)d_in[11];
    const float* W_out     = (const float*)d_in[12];
    const float* b_out     = (const float*)d_in[13];
    const float* W_att_img = (const float*)d_in[14];
    const float* b_att_img = (const float*)d_in[15];
    const float* W_att_h   = (const float*)d_in[16];
    const float* b_att_h   = (const float*)d_in[17];
    const float* v_att     = (const float*)d_in[18];
    const float* b_v_att   = (const float*)d_in[19];
    const float* E         = (const float*)d_in[20];

    float* preds  = (float*)d_out;                                  // [B,T,V]
    float* alphas = preds + (long)BATCH * TSTEPS * VOCAB;           // [B,T,P]

    // workspace (256B-aligned bump allocator)
    char* wp = (char*)d_ws;
    auto alloc = [&](size_t bytes) -> void* {
        void* p = wp; wp += (bytes + 255) & ~(size_t)255; return p;
    };
    bf16_t*   img_bf     = (bf16_t*)  alloc((size_t)BATCH * PPIX * DDIM * 2);
    float*    att_img_f  = (float*)   alloc((size_t)BATCH * PPIX * HDIM * 4);
    unsigned* Wc_p       = (unsigned*)alloc((size_t)(XDIM / 2) * (4 * HDIM) * 4);
    unsigned* Wout_p     = (unsigned*)alloc((size_t)(HDIM / 2) * VPAD * 4);
    unsigned* Wfbeta_p   = (unsigned*)alloc((size_t)(HDIM / 2) * DDIM * 4);
    unsigned* Watth_p    = (unsigned*)alloc((size_t)(HDIM / 2) * HDIM * 4);
    unsigned* Wattimg_p  = (unsigned*)alloc((size_t)(DDIM / 2) * HDIM * 4);
    unsigned* Winith_p   = (unsigned*)alloc((size_t)(DDIM / 2) * HDIM * 4);
    unsigned* Winitc_p   = (unsigned*)alloc((size_t)(DDIM / 2) * HDIM * 4);
    bf16_t*   avg_bf     = (bf16_t*)  alloc((size_t)BATCH * DDIM * 2);
    bf16_t*   emb_bf     = (bf16_t*)  alloc((size_t)BATCH * CAPLEN * HDIM * 2);
    float*    h_f        = (float*)   alloc((size_t)BATCH * HDIM * 4);
    bf16_t*   h_bf       = (bf16_t*)  alloc((size_t)BATCH * HDIM * 2);
    float*    c_f        = (float*)   alloc((size_t)BATCH * HDIM * 4);
    float*    hW_f       = (float*)   alloc((size_t)BATCH * HDIM * 4);
    float*    gate_f     = (float*)   alloc((size_t)BATCH * DDIM * 4);
    float*    ctx_f      = (float*)   alloc((size_t)BATCH * DDIM * 4);
    bf16_t*   x_bf       = (bf16_t*)  alloc((size_t)BATCH * XDIM * 2);
    float*    gates_f    = (float*)   alloc((size_t)BATCH * 4 * HDIM * 4);
    float*    bcomb_f    = (float*)   alloc((size_t)4 * HDIM * 4);

    auto cvt = [&](const float* s, bf16_t* d, long n) {
        int blocks = (int)((n + 255) / 256); if (blocks > 4096) blocks = 4096;
        cvt_f32_bf16<<<blocks, 256, 0, stream>>>(s, d, n);
    };
    auto pack = [&](const float* s1, int rows1, const float* s2, unsigned* d,
                    int Ktot, int Nsrc, int Npad) {
        const long n = (long)(Ktot / 2) * Npad;
        int blocks = (int)((n + 255) / 256); if (blocks > 4096) blocks = 4096;
        cvt_pack_bf16<<<blocks, 256, 0, stream>>>(s1, rows1, s2, d, Ktot, Nsrc, Npad, n);
    };

    // ---- precompute: conversions / packing / pooling / gathers ----
    cvt(img, img_bf, (long)BATCH * PPIX * DDIM);
    pack(Wih, HDIM + DDIM, Whh, Wc_p,     XDIM, 4 * HDIM, 4 * HDIM); // [Wih;Whh]
    pack(W_out,     HDIM, W_out,     Wout_p,    HDIM, VOCAB, VPAD);
    pack(W_fbeta,   HDIM, W_fbeta,   Wfbeta_p,  HDIM, DDIM,  DDIM);
    pack(W_att_h,   HDIM, W_att_h,   Watth_p,   HDIM, HDIM,  HDIM);
    pack(W_att_img, DDIM, W_att_img, Wattimg_p, DDIM, HDIM,  HDIM);
    pack(W_init_h,  DDIM, W_init_h,  Winith_p,  DDIM, HDIM,  HDIM);
    pack(W_init_c,  DDIM, W_init_c,  Winitc_p,  DDIM, HDIM,  HDIM);

    avgpool_kernel<<<(BATCH * DDIM) / 256, 256, 0, stream>>>(img, avg_bf);
    gather_emb<<<(BATCH * CAPLEN * HDIM + 255) / 256, 256, 0, stream>>>(
        captions, E, emb_bf, (long)BATCH * CAPLEN * HDIM);
    add_bias2<<<(4 * HDIM + 255) / 256, 256, 0, stream>>>(bih, bhh, bcomb_f, 4 * HDIM);

    // h0 = tanh(avg @ W_init_h + b) ; c0 = tanh(avg @ W_init_c + b)
    wmma_gemm_bf16<<<dim3(4, 2), 256, 0, stream>>>(
        avg_bf, DDIM, Winith_p, HDIM, b_init_h, h_f, HDIM, BATCH, DDIM, HDIM, 1);
    cvt_f32_bf16<<<64, 256, 0, stream>>>(h_f, h_bf, (long)BATCH * HDIM);
    wmma_gemm_bf16<<<dim3(4, 2), 256, 0, stream>>>(
        avg_bf, DDIM, Winitc_p, HDIM, b_init_c, c_f, HDIM, BATCH, DDIM, HDIM, 1);

    // att_img = img @ W_att_img + b   (6272 x 2048 x 512)
    wmma_gemm_bf16<<<dim3(4, (BATCH * PPIX) / 16), 256, 0, stream>>>(
        img_bf, DDIM, Wattimg_p, HDIM, b_att_img, att_img_f, HDIM,
        BATCH * PPIX, DDIM, HDIM, 0);

    // ---- recurrent timesteps ----
    for (int t = 0; t < TSTEPS; ++t) {
        // hW = h @ W_att_h + b_att_h
        wmma_gemm_bf16<<<dim3(4, 2), 256, 0, stream>>>(
            h_bf, HDIM, Watth_p, HDIM, b_att_h, hW_f, HDIM, BATCH, HDIM, HDIM, 0);

        attention_kernel<<<BATCH, 256, 0, stream>>>(
            att_img_f, hW_f, v_att, b_v_att, img_bf, ctx_f, alphas, t);

        // gate = sigmoid(h @ W_fbeta + b_fbeta)
        wmma_gemm_bf16<<<dim3(16, 2), 256, 0, stream>>>(
            h_bf, HDIM, Wfbeta_p, DDIM, b_fbeta, gate_f, DDIM, BATCH, HDIM, DDIM, 2);

        build_x<<<(BATCH * XDIM) / 256, 256, 0, stream>>>(
            emb_bf, gate_f, ctx_f, h_bf, x_bf, t);

        // gates = [emb, gate*ctx, h] @ [Wih;Whh] + (bih+bhh)
        wmma_gemm_bf16<<<dim3(16, 2), 256, 0, stream>>>(
            x_bf, XDIM, Wc_p, 4 * HDIM, bcomb_f, gates_f, 4 * HDIM,
            BATCH, XDIM, 4 * HDIM, 0);

        lstm_pointwise<<<(BATCH * HDIM) / 256, 256, 0, stream>>>(
            gates_f, c_f, h_f, h_bf);

        // preds[:, t, :] = h @ W_out + b_out   (VPAD = 30080 -> 235 full slabs)
        wmma_gemm_bf16<<<dim3(235, 2), 256, 0, stream>>>(
            h_bf, HDIM, Wout_p, VPAD, b_out, preds + (long)t * VOCAB,
            (long)TSTEPS * VOCAB, BATCH, HDIM, VOCAB, 0);
    }
}
